// FullAttention_44315472560729
// MI455X (gfx1250) — compile-verified, matrix operand order checked by reading, not run on hardware
//
#include <hip/hip_runtime.h>

// MI455X / gfx1250 wave32 WMMA attention.
// One block per (b,t,h): 512 blocks x 256 threads (8 waves).
// HBM-bound (AI ~64 FLOP/B << machine balance): single pass over Q/K/V in HBM,
// f16 WMMA with f32 accumulation, all reuse via LDS. Softmax max is obtained by
// a cheap recompute pass (extra WMMAs from LDS are free vs HBM) so we never
// hold 16 score tiles live -> no VGPR spills.

typedef __attribute__((ext_vector_type(16))) _Float16 v16h;
typedef __attribute__((ext_vector_type(8)))  _Float16 v8h;
typedef __attribute__((ext_vector_type(4)))  _Float16 v4h;
typedef __attribute__((ext_vector_type(8)))  float    v8f;
typedef __attribute__((ext_vector_type(4)))  float    v4f;

constexpr int Bc = 4, Tc = 16, Nc = 256, Hc = 8, DKc = 64;
constexpr int ROWSTRIDE = Hc * DKc;          // 512 floats between n (or s) rows
constexpr float SCALE_LOG2E = 0.125f * 1.44269504088896340736f; // (1/sqrt(64))*log2(e)

// LDS: K tile [256][64] f16, V tile transposed [64][256] f16. 32KB + 32KB = 64KB.
// XOR swizzle at 16B-chunk granularity for conflict-free b128 row reads.
__device__ __forceinline__ int kidx(int s, int halfcol) {
    // row stride 64 halfs (128B = 8 chunks). swizzle: chunk ^ ((s>>1)&7)
    int chunk = halfcol >> 3, off = halfcol & 7;
    return s * 64 + ((chunk ^ ((s >> 1) & 7)) << 3) + off;
}
__device__ __forceinline__ int vidx(int d, int s) {
    // row stride 256 halfs (512B = 32 chunks). swizzle low 4 chunk bits by d&15
    int chunk = s >> 3, off = s & 7;
    return d * 256 + ((chunk ^ (d & 15)) << 3) + off;
}

// Load one 16x32 f16 A-tile row-fragment from LDS: two b128 loads.
__device__ __forceinline__ v16h ld_tile(const _Float16* p0, const _Float16* p1) {
    v8h lo = *(const v8h*)p0;
    v8h hi = *(const v8h*)p1;
    v16h a;
    #pragma unroll
    for (int i = 0; i < 8; ++i) { a[i] = lo[i]; a[8 + i] = hi[i]; }
    return a;
}

__global__ __launch_bounds__(256)
void fullattn_wmma_kernel(const float* __restrict__ Q,
                          const float* __restrict__ K,
                          const float* __restrict__ V,
                          float* __restrict__ O)
{
    __shared__ _Float16 sK[256 * 64];   // [s][e], swizzled
    __shared__ _Float16 sVt[64 * 256];  // [d][s], swizzled

    const int tid  = threadIdx.x;
    const int lane = tid & 31;
    const int wave = tid >> 5;

    const int h  = blockIdx.x % Hc;
    const int bt = blockIdx.x / Hc;
    const size_t base = (size_t)bt * (Nc * ROWSTRIDE) + (size_t)h * DKc;

    const float* Kb = K + base;
    const float* Vb = V + base;
    const float* Qb = Q + base;
    float*       Ob = O + base;

    // ---- stage K: f32 global -> f16 LDS [s][e] (coalesced b128 loads) ----
    #pragma unroll
    for (int it = 0; it < 16; ++it) {
        int idx  = tid + it * 256;       // 4096 float4 units over 256x64
        int s    = idx >> 4;
        int c4   = idx & 15;             // float4 column
        v4f g = *(const v4f*)(Kb + (size_t)s * ROWSTRIDE + c4 * 4);
        v4h hh;
        hh.x = (_Float16)g.x; hh.y = (_Float16)g.y;
        hh.z = (_Float16)g.z; hh.w = (_Float16)g.w;
        *(v4h*)(sK + kidx(s, c4 * 4)) = hh;   // 8B aligned within one 16B chunk
    }
    // ---- stage V transposed: (s,d) global -> LDS [d][s] ----
    #pragma unroll
    for (int it = 0; it < 16; ++it) {
        int idx = tid + it * 256;
        int s   = idx >> 4;
        int d   = (idx & 15) << 2;
        v4f g = *(const v4f*)(Vb + (size_t)s * ROWSTRIDE + d);
        sVt[vidx(d + 0, s)] = (_Float16)g.x;
        sVt[vidx(d + 1, s)] = (_Float16)g.y;
        sVt[vidx(d + 2, s)] = (_Float16)g.z;
        sVt[vidx(d + 3, s)] = (_Float16)g.w;
    }
    __syncthreads();

    // Per the f16 WMMA layouts, A rows and B columns both hold K-dim values
    // {klo..klo+7, klo+16..klo+23}, klo = 0 (lanes 0-15) or 8 (lanes 16-31).
    const int klo = (lane < 16) ? 0 : 8;
    const int lr  = lane & 15;

    #pragma unroll 1
    for (int qt = wave; qt < 16; qt += 8) {       // 2 q-tiles of 16 queries per wave
        const int q0 = qt * 16;

        // Q as B-matrix tiles (column q = lane's Q row), two 32-deep e-chunks
        v16h qb0, qb1;
        {
            const float* qrow = Qb + (size_t)(q0 + lr) * ROWSTRIDE;
            #pragma unroll
            for (int c = 0; c < 2; ++c) {
                const int e0 = 32 * c + klo;
                v4f a0 = *(const v4f*)(qrow + e0);
                v4f a1 = *(const v4f*)(qrow + e0 + 4);
                v4f b0 = *(const v4f*)(qrow + e0 + 16);
                v4f b1 = *(const v4f*)(qrow + e0 + 20);
                v16h t;
                t[0]=(_Float16)a0.x; t[1]=(_Float16)a0.y; t[2]=(_Float16)a0.z; t[3]=(_Float16)a0.w;
                t[4]=(_Float16)a1.x; t[5]=(_Float16)a1.y; t[6]=(_Float16)a1.z; t[7]=(_Float16)a1.w;
                t[8]=(_Float16)b0.x; t[9]=(_Float16)b0.y; t[10]=(_Float16)b0.z; t[11]=(_Float16)b0.w;
                t[12]=(_Float16)b1.x; t[13]=(_Float16)b1.y; t[14]=(_Float16)b1.z; t[15]=(_Float16)b1.w;
                if (c == 0) qb0 = t; else qb1 = t;
            }
        }

        // ---- pass 1: row max of scores^T (tiles are transient; no big live set)
        float m = -3.4e38f;
        #pragma unroll 1
        for (int st = 0; st < 16; ++st) {
            const int srow = st * 16 + lr;
            v8f acc = {};
            v16h a0 = ld_tile(sK + kidx(srow, klo),      sK + kidx(srow, klo + 16));
            acc = __builtin_amdgcn_wmma_f32_16x16x32_f16(false, a0, false, qb0,
                                                         (short)0, acc, false, false);
            v16h a1 = ld_tile(sK + kidx(srow, 32 + klo), sK + kidx(srow, 32 + klo + 16));
            acc = __builtin_amdgcn_wmma_f32_16x16x32_f16(false, a1, false, qb1,
                                                         (short)0, acc, false, false);
            #pragma unroll
            for (int i = 0; i < 8; ++i) m = fmaxf(m, acc[i]);
        }
        m = fmaxf(m, __shfl_xor(m, 16, 32));      // lane pair (L, L^16) owns column q
        const float mm = m * SCALE_LOG2E;

        // ---- pass 2: recompute scores per 32-wide s-chunk, exp, P*V on the fly
        v8f oacc[4] = {v8f{}, v8f{}, v8f{}, v8f{}};
        float sum = 0.f;
        #pragma unroll 1
        for (int scp = 0; scp < 8; ++scp) {
            // two consecutive score^T tiles (s = 32*scp .. 32*scp+31)
            v8f t0 = {}, t1 = {};
            {
                const int r0 = (2 * scp) * 16 + lr;
                v16h a0 = ld_tile(sK + kidx(r0, klo),      sK + kidx(r0, klo + 16));
                t0 = __builtin_amdgcn_wmma_f32_16x16x32_f16(false, a0, false, qb0,
                                                            (short)0, t0, false, false);
                v16h a1 = ld_tile(sK + kidx(r0, 32 + klo), sK + kidx(r0, 32 + klo + 16));
                t0 = __builtin_amdgcn_wmma_f32_16x16x32_f16(false, a1, false, qb1,
                                                            (short)0, t0, false, false);
            }
            {
                const int r1 = (2 * scp + 1) * 16 + lr;
                v16h a0 = ld_tile(sK + kidx(r1, klo),      sK + kidx(r1, klo + 16));
                t1 = __builtin_amdgcn_wmma_f32_16x16x32_f16(false, a0, false, qb0,
                                                            (short)0, t1, false, false);
                v16h a1 = ld_tile(sK + kidx(r1, 32 + klo), sK + kidx(r1, 32 + klo + 16));
                t1 = __builtin_amdgcn_wmma_f32_16x16x32_f16(false, a1, false, qb1,
                                                            (short)0, t1, false, false);
            }
            // exponentiate and pack directly into the f16 B-layout of P
            // (score D-layout -> P B-layout needs zero cross-lane movement)
            v16h pb;
            #pragma unroll
            for (int i = 0; i < 8; ++i) {
                float e0 = exp2f(t0[i] * SCALE_LOG2E - mm);
                float e1 = exp2f(t1[i] * SCALE_LOG2E - mm);
                sum += e0; sum += e1;
                pb[i]     = (_Float16)e0;
                pb[8 + i] = (_Float16)e1;
            }
            // O^T accumulate: 4 d-tiles, A = V^T rows from LDS
            #pragma unroll
            for (int dm = 0; dm < 4; ++dm) {
                const int drow = dm * 16 + lr;
                const int s0   = scp * 32 + klo;
                v16h a = ld_tile(sVt + vidx(drow, s0), sVt + vidx(drow, s0 + 16));
                oacc[dm] = __builtin_amdgcn_wmma_f32_16x16x32_f16(
                               false, a, false, pb, (short)0, oacc[dm], false, false);
            }
        }
        sum += __shfl_xor(sum, 16, 32);
        const float rs = 1.0f / sum;

        // store: lane owns query q0+lr, d = dm*16 + klo + {0..7} -> two float4s
        float* orow = Ob + (size_t)(q0 + lr) * ROWSTRIDE;
        #pragma unroll
        for (int dm = 0; dm < 4; ++dm) {
            v4f f0, f1;
            f0.x = oacc[dm][0] * rs; f0.y = oacc[dm][1] * rs;
            f0.z = oacc[dm][2] * rs; f0.w = oacc[dm][3] * rs;
            f1.x = oacc[dm][4] * rs; f1.y = oacc[dm][5] * rs;
            f1.z = oacc[dm][6] * rs; f1.w = oacc[dm][7] * rs;
            const int dbase = dm * 16 + klo;
            *(v4f*)(orow + dbase)     = f0;
            *(v4f*)(orow + dbase + 4) = f1;
        }
    }
}

extern "C" void kernel_launch(void* const* d_in, const int* in_sizes, int n_in,
                              void* d_out, int out_size, void* d_ws, size_t ws_size,
                              hipStream_t stream) {
    (void)in_sizes; (void)n_in; (void)d_ws; (void)ws_size; (void)out_size;
    const float* Q = (const float*)d_in[0];
    const float* K = (const float*)d_in[1];
    const float* V = (const float*)d_in[2];
    // d_in[3] attention_mask unused (mask_flag=False in reference)
    float* O = (float*)d_out;

    dim3 grid(Bc * Tc * Hc);   // 512 blocks, one per (b,t,h)
    dim3 block(256);           // 8 wave32 waves
    fullattn_wmma_kernel<<<grid, block, 0, stream>>>(Q, K, V, O);
}
